// CloudSecurityGNN_72121090835130
// MI455X (gfx1250) — matrix operand order repaired and checked by reading, not compiled, not run on hardware
//
#include <hip/hip_runtime.h>
#include <hip/hip_bf16.h>

// ---------------- problem constants ----------------
#define N_NODES 50000
#define N_EDGES 800000
#define EP      (N_EDGES + N_NODES)   // edges + self loops = 850000
#define F_IN0   128
#define HC      256                   // H * C
#define C_CH    64
#define G_GR    64

typedef __attribute__((ext_vector_type(16))) __bf16 bf16x16;
typedef __attribute__((ext_vector_type(8)))  __bf16 bf16x8;
typedef __attribute__((ext_vector_type(8)))  float  f32x8;

// ---------------- helpers ----------------
__device__ __forceinline__ float wave_red_sum(float v) {
  #pragma unroll
  for (int o = 16; o; o >>= 1) v += __shfl_xor(v, o, 32);
  return v;
}

__global__ void fill_u32(unsigned int* p, unsigned int v, int n) {
  int t = blockIdx.x * blockDim.x + threadIdx.x;
  if (t < n) p[t] = v;
}

__global__ void f32_to_bf16_k(const float* __restrict__ in, __bf16* __restrict__ out, int n) {
  int t = blockIdx.x * blockDim.x + threadIdx.x;
  if (t < n) out[t] = (__bf16)in[t];
}

// W [K x 256] f32  ->  Wt [256 x K] bf16
__global__ void transpose_w_bf16(const float* __restrict__ W, __bf16* __restrict__ Wt, int K) {
  int t = blockIdx.x * blockDim.x + threadIdx.x;
  if (t >= K * 256) return;
  int k = t >> 8, oc = t & 255;
  Wt[(size_t)oc * K + k] = (__bf16)W[t];
}

// ---------------- WMMA GEMM: h = A(bf16 NxK) @ Bt^T (bf16 256xK) -> f32 Nx256 ----
// block = 128 threads = 4 waves; wave w -> cols [64w, 64w+64); grid.x = N/16
__global__ __launch_bounds__(128)
void gemm_bf16_wmma(const __bf16* __restrict__ A, const __bf16* __restrict__ Bt,
                    float* __restrict__ Cm, int K) {
  const int lane  = threadIdx.x & 31;
  const int wave  = threadIdx.x >> 5;
  const int row0  = blockIdx.x * 16;
  const int row   = row0 + (lane & 15);
  const int halfA = (lane >> 4) ? 8 : 0;    // ISA 16-bit A layout: hi lanes hold K+8
  const int halfB = (lane >> 4) ? 8 : 0;    // ISA 16-bit B layout: hi lanes hold K+8
  const int colb  = wave * 64;

  f32x8 acc0 = {}, acc1 = {}, acc2 = {}, acc3 = {};

  for (int kb = 0; kb < K; kb += 32) {
    bf16x16 afrag;
    {
      const __bf16* ap = A + (size_t)row * K + kb + halfA;
      bf16x8 lo = *(const bf16x8*)(ap);        // K = half .. half+7
      bf16x8 hi = *(const bf16x8*)(ap + 16);   // K = half+16 .. half+23
      #pragma unroll
      for (int i = 0; i < 8; i++) { afrag[i] = lo[i]; afrag[i + 8] = hi[i]; }
    }
    #pragma unroll
    for (int t = 0; t < 4; t++) {
      const int col = colb + t * 16 + (lane & 15);
      const __bf16* bp = Bt + (size_t)col * K + kb + halfB;
      bf16x16 bfrag;
      bf16x8 lo = *(const bf16x8*)(bp);
      bf16x8 hi = *(const bf16x8*)(bp + 16);
      #pragma unroll
      for (int i = 0; i < 8; i++) { bfrag[i] = lo[i]; bfrag[i + 8] = hi[i]; }
      f32x8& acc = (t == 0) ? acc0 : (t == 1) ? acc1 : (t == 2) ? acc2 : acc3;
      acc = __builtin_amdgcn_wmma_f32_16x16x32_bf16(false, afrag, false, bfrag,
                                                    (short)0, acc, false, false);
    }
  }

  const int mrow0 = row0 + ((lane >> 4) ? 8 : 0);  // ISA C/D layout
  #pragma unroll
  for (int t = 0; t < 4; t++) {
    const f32x8& acc = (t == 0) ? acc0 : (t == 1) ? acc1 : (t == 2) ? acc2 : acc3;
    const int col = colb + t * 16 + (lane & 15);
    #pragma unroll
    for (int r = 0; r < 8; r++)
      Cm[(size_t)(mrow0 + r) * HC + col] = acc[r];
  }
}

// ---------------- attention logits: als/ald [N,4] ----------------
__global__ void attn_logits(const float* __restrict__ hmat, const float* __restrict__ asrc,
                            const float* __restrict__ adst, float* __restrict__ als,
                            float* __restrict__ ald) {
  int t = blockIdx.x * blockDim.x + threadIdx.x;
  if (t >= N_NODES * 4) return;
  int n = t >> 2, h = t & 3;
  const float* hp = hmat + (size_t)n * HC + h * 64;
  const float* ap = asrc + h * 64;
  const float* bp = adst + h * 64;
  float ss = 0.f, sd = 0.f;
  #pragma unroll 4
  for (int c = 0; c < 64; c++) { float x = hp[c]; ss += x * ap[c]; sd += x * bp[c]; }
  als[t] = ss; ald[t] = sd;
}

// ---------------- pass 1: e = leakyrelu(als[src]+ald[dst]); segment max ----------
__global__ void edge_logit(const int* __restrict__ ei, const float* __restrict__ als,
                           const float* __restrict__ ald, float* __restrict__ ebuf,
                           int* __restrict__ mmax) {
  int e = blockIdx.x * blockDim.x + threadIdx.x;
  if (e >= EP) return;
  int s, d;
  if (e < N_EDGES) { s = ei[e]; d = ei[N_EDGES + e]; } else { s = d = e - N_EDGES; }
  #pragma unroll
  for (int h = 0; h < 4; h++) {
    float v = als[s * 4 + h] + ald[d * 4 + h];
    v = v > 0.f ? v : 0.2f * v;
    ebuf[e * 4 + h] = v;
    if (v >= 0.f) atomicMax((int*)&mmax[d * 4 + h], __float_as_int(v));
    else          atomicMin((unsigned int*)&mmax[d * 4 + h], __float_as_uint(v));
  }
}

// ---------------- pass 2: p = exp(e - m[dst]); segment sum ----------------
__global__ void edge_exp(const int* __restrict__ ei, const int* __restrict__ mmax,
                         float* __restrict__ ebuf, float* __restrict__ den) {
  int e = blockIdx.x * blockDim.x + threadIdx.x;
  if (e >= EP) return;
  int d = (e < N_EDGES) ? ei[N_EDGES + e] : e - N_EDGES;
  #pragma unroll
  for (int h = 0; h < 4; h++) {
    float p = __expf(ebuf[e * 4 + h] - __int_as_float(mmax[d * 4 + h]));
    ebuf[e * 4 + h] = p;
    unsafeAtomicAdd(&den[d * 4 + h], p);
  }
}

// ---------------- pass 3: out[dst] += alpha * h[src]; one wave per edge ---------
__global__ __launch_bounds__(256)
void edge_msg(const int* __restrict__ ei, const float* __restrict__ ebuf,
              const float* __restrict__ den, const float* __restrict__ hmat,
              float* __restrict__ outacc) {
  int wid  = (blockIdx.x * 256 + threadIdx.x) >> 5;
  int lane = threadIdx.x & 31;
  if (wid >= EP) return;
  int s, d;
  if (wid < N_EDGES) { s = ei[wid]; d = ei[N_EDGES + wid]; } else { s = d = wid - N_EDGES; }
  const float* hr = hmat + (size_t)s * HC;
  float* orow = outacc + (size_t)d * HC;
  __builtin_prefetch(hr, 0, 1);
  float al[4];
  #pragma unroll
  for (int h = 0; h < 4; h++) al[h] = ebuf[wid * 4 + h] / den[d * 4 + h];
  #pragma unroll
  for (int i = 0; i < 8; i++) {           // head index (i*32+lane)>>6 == i>>1 (static)
    int idx = i * 32 + lane;
    unsafeAtomicAdd(&orow[idx], al[i >> 1] * hr[idx]);
  }
}

// ---------------- finalize: head-mean + bias (+relu), emit f32 (+bf16) ----------
__global__ void finalize_layer(const float* __restrict__ acc, const float* __restrict__ bias,
                               float* __restrict__ out32, __bf16* out16, int do_relu) {
  int t = blockIdx.x * blockDim.x + threadIdx.x;
  if (t >= N_NODES * C_CH) return;
  int n = t >> 6, c = t & 63;
  const float* r = acc + (size_t)n * HC;
  float v = (r[c] + r[64 + c] + r[128 + c] + r[192 + c]) * 0.25f + bias[c];
  if (do_relu) v = fmaxf(v, 0.f);
  out32[t] = v;
  if (out16) out16[t] = (__bf16)v;
}

// ---------------- node MLP heads + pooling (wave per node) ----------------
__global__ __launch_bounds__(256)
void node_heads(const float* __restrict__ emb, const int* __restrict__ batch,
                const float* aw1, const float* ab1, const float* aw2, const float* ab2,
                const float* rw1, const float* rb1, const float* rw2, const float* rb2,
                const float* cw1, const float* cb1, const float* cw2, const float* cb2,
                float* __restrict__ anom, float* __restrict__ risk, float* __restrict__ resrc,
                float* __restrict__ psum, float* __restrict__ pcnt) {
  int wid  = (blockIdx.x * 256 + threadIdx.x) >> 5;
  int lane = threadIdx.x & 31;
  if (wid >= N_NODES) return;
  const float* er = emb + (size_t)wid * 64;
  float e0 = er[lane], e1 = er[32 + lane];
  int b = batch[wid];
  unsafeAtomicAdd(&psum[b * 64 + lane], e0);
  unsafeAtomicAdd(&psum[b * 64 + 32 + lane], e1);
  if (lane == 0) unsafeAtomicAdd(&pcnt[b], 1.0f);

  float hA = ab1[lane], hR = rb1[lane], hC = cb1[lane];
  for (int k = 0; k < 64; k++) {
    float ev = (k < 32) ? __shfl(e0, k, 32) : __shfl(e1, k - 32, 32);
    hA += ev * aw1[k * 32 + lane];
    hR += ev * rw1[k * 32 + lane];
    hC += ev * cw1[k * 32 + lane];
  }
  hA = fmaxf(hA, 0.f); hR = fmaxf(hR, 0.f); hC = fmaxf(hC, 0.f);

  float pa = wave_red_sum(hA * aw2[lane]);
  float pr = wave_red_sum(hR * rw2[lane]);
  float pc0 = wave_red_sum(hC * cw2[lane * 5 + 0]);
  float pc1 = wave_red_sum(hC * cw2[lane * 5 + 1]);
  float pc2 = wave_red_sum(hC * cw2[lane * 5 + 2]);
  float pc3 = wave_red_sum(hC * cw2[lane * 5 + 3]);
  float pc4 = wave_red_sum(hC * cw2[lane * 5 + 4]);
  if (lane == 0) {
    anom[wid] = 1.f / (1.f + __expf(-(pa + ab2[0])));
    risk[wid] = 1.f / (1.f + __expf(-(pr + rb2[0])));
    resrc[wid * 5 + 0] = pc0 + cb2[0];
    resrc[wid * 5 + 1] = pc1 + cb2[1];
    resrc[wid * 5 + 2] = pc2 + cb2[2];
    resrc[wid * 5 + 3] = pc3 + cb2[3];
    resrc[wid * 5 + 4] = pc4 + cb2[4];
  }
}

// ---------------- graph head: 64 graphs ----------------
__global__ void graph_head(const float* __restrict__ psum, const float* __restrict__ pcnt,
                           const float* gw1, const float* gb1, const float* gw2, const float* gb2,
                           float* __restrict__ out) {
  int g = threadIdx.x;
  if (g >= G_GR) return;
  float inv = 1.f / fmaxf(pcnt[g], 1.f);
  float hid[32];
  #pragma unroll
  for (int j = 0; j < 32; j++) hid[j] = gb1[j];
  for (int k = 0; k < 64; k++) {
    float pv = psum[g * 64 + k] * inv;
    #pragma unroll
    for (int j = 0; j < 32; j++) hid[j] += pv * gw1[k * 32 + j];
  }
  #pragma unroll
  for (int j = 0; j < 32; j++) hid[j] = fmaxf(hid[j], 0.f);
  #pragma unroll
  for (int o = 0; o < 4; o++) {
    float s = gb2[o];
    #pragma unroll
    for (int j = 0; j < 32; j++) s += hid[j] * gw2[j * 4 + o];
    out[g * 4 + o] = s;
  }
}

// ================= host side =================
static inline size_t alignup(size_t x) { return (x + 255) & ~(size_t)255; }

extern "C" void kernel_launch(void* const* d_in, const int* in_sizes, int n_in,
                              void* d_out, int out_size, void* d_ws, size_t ws_size,
                              hipStream_t stream) {
  (void)in_sizes; (void)n_in; (void)out_size; (void)ws_size;
  const float* x      = (const float*)d_in[0];
  const int*   ei     = (const int*)d_in[1];
  const int*   batch  = (const int*)d_in[2];
  const float* W[3]   = { (const float*)d_in[3], (const float*)d_in[7],  (const float*)d_in[11] };
  const float* as_[3] = { (const float*)d_in[4], (const float*)d_in[8],  (const float*)d_in[12] };
  const float* ad_[3] = { (const float*)d_in[5], (const float*)d_in[9],  (const float*)d_in[13] };
  const float* b_[3]  = { (const float*)d_in[6], (const float*)d_in[10], (const float*)d_in[14] };
  const float* aw1 = (const float*)d_in[15]; const float* ab1 = (const float*)d_in[16];
  const float* aw2 = (const float*)d_in[17]; const float* ab2 = (const float*)d_in[18];
  const float* rw1 = (const float*)d_in[19]; const float* rb1 = (const float*)d_in[20];
  const float* rw2 = (const float*)d_in[21]; const float* rb2 = (const float*)d_in[22];
  const float* cw1 = (const float*)d_in[23]; const float* cb1 = (const float*)d_in[24];
  const float* cw2 = (const float*)d_in[25]; const float* cb2 = (const float*)d_in[26];
  const float* gw1 = (const float*)d_in[27]; const float* gb1 = (const float*)d_in[28];
  const float* gw2 = (const float*)d_in[29]; const float* gb2 = (const float*)d_in[30];

  // workspace bump allocator (ebuf aliases x16: x16 is dead after layer-1 GEMM)
  char* p = (char*)d_ws; size_t off = 0;
  auto grab = [&](size_t bytes) { char* r = p + off; off += alignup(bytes); return r; };
  __bf16* x16    = (__bf16*)grab((size_t)EP * 4 * sizeof(float)); // also ebuf (13.6MB > 12.8MB)
  float*  ebuf   = (float*)x16;
  __bf16* wt[3]  = { (__bf16*)grab(256 * 128 * 2), (__bf16*)grab(256 * 128 * 2),
                     (__bf16*)grab(256 * 128 * 2) };
  float*  hmat   = (float*)grab((size_t)N_NODES * HC * sizeof(float));
  float*  outacc = (float*)grab((size_t)N_NODES * HC * sizeof(float));
  float*  hin32  = (float*)grab((size_t)N_NODES * C_CH * sizeof(float));
  __bf16* hin16  = (__bf16*)grab((size_t)N_NODES * C_CH * 2);
  float*  als    = (float*)grab((size_t)N_NODES * 4 * sizeof(float));
  float*  ald    = (float*)grab((size_t)N_NODES * 4 * sizeof(float));
  int*    mmax   = (int*)grab((size_t)N_NODES * 4 * sizeof(int));
  float*  den    = (float*)grab((size_t)N_NODES * 4 * sizeof(float));
  float*  psum   = (float*)grab((size_t)G_GR * 64 * sizeof(float));
  float*  pcnt   = (float*)grab((size_t)G_GR * sizeof(float));

  float* o_emb  = (float*)d_out;                  // [N,64]
  float* o_anom = o_emb + (size_t)N_NODES * 64;   // [N,1]
  float* o_risk = o_anom + N_NODES;               // [N,1]
  float* o_res  = o_risk + N_NODES;               // [N,5]
  float* o_gr   = o_res + (size_t)N_NODES * 5;    // [G,4]

  // x -> bf16
  f32_to_bf16_k<<<(N_NODES * F_IN0 + 255) / 256, 256, 0, stream>>>(x, x16, N_NODES * F_IN0);

  auto run_layer = [&](const __bf16* inbf, int K, int li, float* out32, __bf16* out16, int relu) {
    transpose_w_bf16<<<(K * 256 + 255) / 256, 256, 0, stream>>>(W[li], wt[li], K);
    gemm_bf16_wmma<<<N_NODES / 16, 128, 0, stream>>>(inbf, wt[li], hmat, K);
    attn_logits<<<(N_NODES * 4 + 255) / 256, 256, 0, stream>>>(hmat, as_[li], ad_[li], als, ald);
    fill_u32<<<(N_NODES * 4 + 255) / 256, 256, 0, stream>>>((unsigned int*)mmax, 0xFF800000u, N_NODES * 4);
    fill_u32<<<(N_NODES * 4 + 255) / 256, 256, 0, stream>>>((unsigned int*)den, 0u, N_NODES * 4);
    fill_u32<<<(N_NODES * HC + 255) / 256, 256, 0, stream>>>((unsigned int*)outacc, 0u, N_NODES * HC);
    edge_logit<<<(EP + 255) / 256, 256, 0, stream>>>(ei, als, ald, ebuf, mmax);
    edge_exp<<<(EP + 255) / 256, 256, 0, stream>>>(ei, mmax, ebuf, den);
    edge_msg<<<(EP * 32 + 255) / 256, 256, 0, stream>>>(ei, ebuf, den, hmat, outacc);
    finalize_layer<<<(N_NODES * C_CH + 255) / 256, 256, 0, stream>>>(outacc, b_[li], out32, out16, relu);
  };

  run_layer(x16,   F_IN0, 0, hin32, hin16, 1);
  run_layer(hin16, C_CH,  1, hin32, hin16, 1);
  run_layer(hin16, C_CH,  2, o_emb, nullptr, 0);   // emb straight into d_out

  fill_u32<<<(G_GR * 64 + 255) / 256, 256, 0, stream>>>((unsigned int*)psum, 0u, G_GR * 64);
  fill_u32<<<1, 64, 0, stream>>>((unsigned int*)pcnt, 0u, G_GR);
  node_heads<<<(N_NODES * 32 + 255) / 256, 256, 0, stream>>>(
      o_emb, batch, aw1, ab1, aw2, ab2, rw1, rb1, rw2, rb2, cw1, cb1, cw2, cb2,
      o_anom, o_risk, o_res, psum, pcnt);
  graph_head<<<1, 64, 0, stream>>>(psum, pcnt, gw1, gb1, gw2, gb2, o_gr);
}
